// MixtureOfExperts_20194936226469
// MI455X (gfx1250) — matrix operand order repaired and checked by reading, not compiled, not run on hardware
//
#include <hip/hip_runtime.h>
#include <hip/hip_bf16.h>

// ---------------- types ----------------
typedef __attribute__((ext_vector_type(8)))  float        v8f;
typedef __attribute__((ext_vector_type(8)))  __bf16       v8bf;
typedef __attribute__((ext_vector_type(16))) __bf16       v16bf;
typedef __attribute__((ext_vector_type(4)))  unsigned int v4u;
typedef __attribute__((ext_vector_type(4)))  int          v4i;
typedef __attribute__((ext_vector_type(8)))  int          v8i;

#define NTOK 4096   // B*T
#define DDIM 1024
#define FDIM 4096
#define NEXP 8
#define TOPK 2

#if defined(__has_builtin)
#if __has_builtin(__builtin_amdgcn_tensor_load_to_lds) && \
    __has_builtin(__builtin_amdgcn_s_wait_tensorcnt)
#define HAVE_TDM 1
#endif
#endif

__device__ __forceinline__ v16bf cat16(v8bf lo, v8bf hi) {
  return __builtin_shufflevector(lo, hi, 0,1,2,3,4,5,6,7,8,9,10,11,12,13,14,15);
}

__device__ __forceinline__ v8f wmma_bf16(v16bf a, v16bf b, v8f c) {
  // D = A(16x32 bf16) * B(32x16 bf16) + C(16x16 f32)
  return __builtin_amdgcn_wmma_f32_16x16x32_bf16(
      false, a, false, b, (short)0, c, false, false);
}

__device__ __forceinline__ float fast_rcp(float x) {
  return __builtin_amdgcn_rcpf(x);   // v_rcp_f32
}

// ---------------- small kernels ----------------
__global__ void moe_init(int* __restrict__ cnt) {
  int i = threadIdx.x;
  if (i < NEXP) cnt[i] = 0;
}

__global__ void moe_cvt_bf16(const float* __restrict__ x, __bf16* __restrict__ xb) {
  size_t i = (size_t)blockIdx.x * blockDim.x + threadIdx.x;
  xb[i] = (__bf16)x[i];
}

// One wave per token: logits = x[t,:] @ Wr, top-2 + softmax weights,
// full softmax probs for the load-balancing loss, scatter to expert lists.
__global__ __launch_bounds__(256) void moe_router(
    const float* __restrict__ x, const float* __restrict__ Wr,
    float* __restrict__ probs, int* __restrict__ cnt,
    int* __restrict__ tok, float* __restrict__ wl) {
  const int tid  = threadIdx.x;
  const int lane = tid & 31;
  const int t    = blockIdx.x * 8 + (tid >> 5);

  float acc[NEXP];
#pragma unroll
  for (int e = 0; e < NEXP; ++e) acc[e] = 0.f;

  const float* xr = x + (size_t)t * DDIM;
  for (int d = lane; d < DDIM; d += 32) {
    float xv = xr[d];
    const float* wr = Wr + (size_t)d * NEXP;
#pragma unroll
    for (int e = 0; e < NEXP; ++e) acc[e] += xv * wr[e];
  }
#pragma unroll
  for (int e = 0; e < NEXP; ++e) {
#pragma unroll
    for (int off = 16; off > 0; off >>= 1)
      acc[e] += __shfl_xor(acc[e], off, 32);
  }

  if (lane == 0) {
    int i0 = 0; float v0 = acc[0];
#pragma unroll
    for (int e = 1; e < NEXP; ++e) if (acc[e] > v0) { v0 = acc[e]; i0 = e; }
    int i1 = -1; float v1 = -1e30f;
#pragma unroll
    for (int e = 0; e < NEXP; ++e)
      if (e != i0 && acc[e] > v1) { v1 = acc[e]; i1 = e; }

    float e1 = __expf(v1 - v0);
    float w0 = fast_rcp(1.f + e1);
    float w1 = e1 * w0;

    float s = 0.f, p[NEXP];
#pragma unroll
    for (int e = 0; e < NEXP; ++e) { p[e] = __expf(acc[e] - v0); s += p[e]; }
    float inv = fast_rcp(s);
#pragma unroll
    for (int e = 0; e < NEXP; ++e) probs[(size_t)t * NEXP + e] = p[e] * inv;

    int p0 = atomicAdd(&cnt[i0], 1);
    tok[i0 * NTOK + p0] = t * TOPK + 0;  wl[i0 * NTOK + p0] = w0;
    int p1 = atomicAdd(&cnt[i1], 1);
    tok[i1 * NTOK + p1] = t * TOPK + 1;  wl[i1 * NTOK + p1] = w1;
  }
}

__global__ void moe_offsets(const int* __restrict__ cnt, int* __restrict__ offs) {
  if (threadIdx.x == 0) {
    int s = 0;
#pragma unroll
    for (int e = 0; e < NEXP; ++e) { offs[e] = s; s += cnt[e]; }
    offs[NEXP] = s;
  }
}

// Deterministic single-block reduction for the aux loss.
__global__ __launch_bounds__(256) void moe_loss(const float* __restrict__ probs,
                                                float* __restrict__ out_loss) {
  __shared__ float red[256];
  float us[NEXP];
#pragma unroll
  for (int e = 0; e < NEXP; ++e) us[e] = 0.f;
  for (int t = threadIdx.x; t < NTOK; t += 256) {
#pragma unroll
    for (int e = 0; e < NEXP; ++e) us[e] += probs[(size_t)t * NEXP + e];
  }
  float loss = 0.f;
  for (int e = 0; e < NEXP; ++e) {
    red[threadIdx.x] = us[e];
    __syncthreads();
    for (int s = 128; s > 0; s >>= 1) {
      if (threadIdx.x < s) red[threadIdx.x] += red[threadIdx.x + s];
      __syncthreads();
    }
    if (threadIdx.x == 0) {
      float u = red[0] * (1.f / (float)NTOK);
      loss += (float)NEXP * u * u;
    }
    __syncthreads();
  }
  if (threadIdx.x == 0) *out_loss = loss;
}

// ---------------- expert GEMM 1: h = silu(x@W1[e]) * (x@W3[e]) ----------------
// Block: 256 thr (8 waves). Tile: 32 tokens x 128 F cols. K loop over D, step 32.
// 4 WMMAs per wave per K-step; b128-vectorized staging and epilogue stores.
__global__ __launch_bounds__(256) void moe_gateup(
    const __bf16* __restrict__ xb,
    const float* __restrict__ W1, const float* __restrict__ W3,
    const int* __restrict__ cnt, const int* __restrict__ offs,
    const int* __restrict__ tok, __bf16* __restrict__ hbuf) {
  const int e   = blockIdx.z;
  const int n_e = cnt[e];
  const int m0  = blockIdx.y * 32;
  if (m0 >= n_e) return;
  const int n0      = blockIdx.x * 128;
  const int rowbase = offs[e] + m0;

  __shared__ __align__(16) __bf16 sA[32][32];
  __shared__ __align__(16) __bf16 sB1[128][32];   // transposed: [n][k]
  __shared__ __align__(16) __bf16 sB3[128][32];
  __shared__ int sTok[32];

  const int tid  = threadIdx.x;
  const int lane = tid & 31;
  const int wv   = tid >> 5;     // wave -> 16-col slice
  const int g    = lane >> 4;    // lane half
  const int ml   = lane & 15;

  if (tid < 32) sTok[tid] = (m0 + tid < n_e) ? (tok[e * NTOK + m0 + tid] >> 1) : -1;
  __syncthreads();

  const float* W1e = W1 + (size_t)e * DDIM * FDIM;
  const float* W3e = W3 + (size_t)e * DDIM * FDIM;

  v8f accG0 = {}, accG1 = {}, accU0 = {}, accU1 = {};

  for (int kk = 0; kk < DDIM; kk += 32) {
    if (kk + 32 < DDIM) {
      size_t pf = (size_t)(kk + 32 + (tid >> 3)) * FDIM + n0 + (tid & 7) * 16;
      __builtin_prefetch((const void*)(W1e + pf), 0, 1);
      __builtin_prefetch((const void*)(W3e + pf), 0, 1);
    }
    // A tile: gathered token rows, 32x32 bf16, 16B chunks
    if (tid < 128) {
      int r = tid >> 2, c = tid & 3;
      int trow = sTok[r];
      if (trow >= 0)
        *(int4*)&sA[r][c * 8] = *(const int4*)(xb + (size_t)trow * DDIM + kk + c * 8);
      else
        *(int4*)&sA[r][c * 8] = make_int4(0, 0, 0, 0);
    }
    // B tiles: 32(k) x 128(n), float4 global loads, transposed bf16 LDS stores
    for (int idx = tid; idx < 32 * 32; idx += 256) {
      int r = idx >> 5, c4 = idx & 31;
      size_t src = (size_t)(kk + r) * FDIM + n0 + c4 * 4;
      float4 a = *(const float4*)(W1e + src);
      float4 b = *(const float4*)(W3e + src);
      int n = c4 * 4;
      sB1[n + 0][r] = (__bf16)a.x; sB1[n + 1][r] = (__bf16)a.y;
      sB1[n + 2][r] = (__bf16)a.z; sB1[n + 3][r] = (__bf16)a.w;
      sB3[n + 0][r] = (__bf16)b.x; sB3[n + 1][r] = (__bf16)b.y;
      sB3[n + 2][r] = (__bf16)b.z; sB3[n + 3][r] = (__bf16)b.w;
    }
    __syncthreads();

    v16bf a0 = cat16(*(const v8bf*)&sA[ml][g * 8],
                     *(const v8bf*)&sA[ml][16 + g * 8]);
    v16bf a1 = cat16(*(const v8bf*)&sA[16 + ml][g * 8],
                     *(const v8bf*)&sA[16 + ml][16 + g * 8]);
    const int nc = wv * 16 + ml;
    v16bf b1 = cat16(*(const v8bf*)&sB1[nc][g * 8],
                     *(const v8bf*)&sB1[nc][16 + g * 8]);
    v16bf b3 = cat16(*(const v8bf*)&sB3[nc][g * 8],
                     *(const v8bf*)&sB3[nc][16 + g * 8]);

    accG0 = wmma_bf16(a0, b1, accG0);
    accG1 = wmma_bf16(a1, b1, accG1);
    accU0 = wmma_bf16(a0, b3, accU0);
    accU1 = wmma_bf16(a1, b3, accU1);
    __syncthreads();
  }

  // epilogue: h = silu(gate)*up -> repack in LDS -> b128 stores
  __bf16 (*sOut)[128] = (__bf16(*)[128])(&sB1[0][0]);   // reuse 8KB
  const int colLoc = wv * 16 + ml;
#pragma unroll
  for (int sub = 0; sub < 2; ++sub) {
#pragma unroll
    for (int r = 0; r < 8; ++r) {
      int M = sub * 16 + r + g * 8;
      float gt = sub ? accG1[r] : accG0[r];
      float up = sub ? accU1[r] : accU0[r];
      float h  = gt * fast_rcp(1.f + __expf(-gt)) * up;
      sOut[M][colLoc] = (__bf16)h;
    }
  }
  __syncthreads();
  for (int idx = tid; idx < 32 * 16; idx += 256) {   // 32 rows x 16 16B-chunks
    int row = idx >> 4, ch = idx & 15;
    if (m0 + row < n_e)
      *(int4*)(hbuf + (size_t)(rowbase + row) * FDIM + n0 + ch * 8) =
          *(const int4*)&sOut[row][ch * 8];
  }
}

// ---------------- expert GEMM 2: partial[slot] = w * (h @ W2[e]) ----------------
// Block: 256 thr. Tile: 32 rows x 128 D cols. K loop over F, step 32.
// A tile loaded by the Tensor Data Mover when available (2D tile, 32x32 bf16).
__global__ __launch_bounds__(256) void moe_down(
    const __bf16* __restrict__ hbuf, const float* __restrict__ W2,
    const int* __restrict__ cnt, const int* __restrict__ offs,
    const int* __restrict__ tok, const float* __restrict__ wl,
    float* __restrict__ partial) {
  const int e   = blockIdx.z;
  const int n_e = cnt[e];
  const int m0  = blockIdx.y * 32;
  if (m0 >= n_e) return;
  const int n0      = blockIdx.x * 128;
  const int rowbase = offs[e] + m0;

  __shared__ __align__(16) __bf16 sA[32][32];
  __shared__ __align__(16) __bf16 sB[128][32];   // transposed: [n][k]
  __shared__ __align__(16) float  sOutF[32][128];
  __shared__ int   sSlot[32];
  __shared__ float sW[32];

  const int tid  = threadIdx.x;
  const int lane = tid & 31;
  const int wv   = tid >> 5;
  const int g    = lane >> 4;
  const int ml   = lane & 15;

  if (tid < 32) {
    bool v = (m0 + tid) < n_e;
    int li = e * NTOK + m0 + tid;
    sSlot[tid] = v ? tok[li] : -1;
    sW[tid]    = v ? wl[li]  : 0.f;
  }
  __syncthreads();

  const float* W2e = W2 + (size_t)e * FDIM * DDIM;

#if HAVE_TDM
  // Invariant parts of the D# for the 32x32 bf16 A tile of hbuf.
  const unsigned ldsA = (unsigned)(uintptr_t)(void*)sA;   // LDS byte offset
  // group1: data_size=2B; tensor_dim0=FDIM, tensor_dim1=2*NTOK;
  // tile 32x32; tensor_dim0_stride=FDIM
  const v8i g1 = { 0x10000,                 // data_size=1 (2 bytes)
                   (int)((FDIM & 0xFFFF) << 16),
                   (int)(((2 * NTOK) & 0xFFFF) << 16),
                   32 << 16,                // tile_dim0 = 32
                   32,                      // tile_dim1 = 32
                   FDIM,                    // tensor_dim0_stride lo
                   0, 0 };
  const v4i z4 = { 0, 0, 0, 0 };
#endif

  v8f acc0 = {}, acc1 = {};

  for (int kk = 0; kk < FDIM; kk += 32) {
#if HAVE_TDM
    if (wv == 0) {
      unsigned long long gaddr =
          (unsigned long long)(uintptr_t)(hbuf + (size_t)rowbase * FDIM + kk);
      v4u g0;
      g0[0] = 1u;                                         // count=1
      g0[1] = ldsA;                                       // lds_addr
      g0[2] = (unsigned)gaddr;                            // global_addr lo
      g0[3] = (unsigned)((gaddr >> 32) & 0x01FFFFFFu) | (2u << 30); // hi | type=2
#if __clang_major__ >= 23
      const v8i z8 = { 0, 0, 0, 0, 0, 0, 0, 0 };
      __builtin_amdgcn_tensor_load_to_lds(g0, g1, z4, z4, z8, 0);
#else
      __builtin_amdgcn_tensor_load_to_lds(g0, g1, z4, z4, 0);
#endif
    }
#else
    if (tid < 128) {
      int r = tid >> 2, c = tid & 3;
      *(int4*)&sA[r][c * 8] =
          *(const int4*)(hbuf + (size_t)(rowbase + r) * FDIM + kk + c * 8);
    }
#endif
    if (kk + 32 < FDIM) {
      size_t pf = (size_t)(kk + 32 + (tid >> 3)) * DDIM + n0 + (tid & 7) * 16;
      __builtin_prefetch((const void*)(W2e + pf), 0, 1);
    }
    for (int idx = tid; idx < 32 * 32; idx += 256) {
      int r = idx >> 5, c4 = idx & 31;
      float4 b = *(const float4*)(W2e + (size_t)(kk + r) * DDIM + n0 + c4 * 4);
      int n = c4 * 4;
      sB[n + 0][r] = (__bf16)b.x; sB[n + 1][r] = (__bf16)b.y;
      sB[n + 2][r] = (__bf16)b.z; sB[n + 3][r] = (__bf16)b.w;
    }
#if HAVE_TDM
    if (wv == 0) __builtin_amdgcn_s_wait_tensorcnt(0);
#endif
    __syncthreads();

    v16bf a0 = cat16(*(const v8bf*)&sA[ml][g * 8],
                     *(const v8bf*)&sA[ml][16 + g * 8]);
    v16bf a1 = cat16(*(const v8bf*)&sA[16 + ml][g * 8],
                     *(const v8bf*)&sA[16 + ml][16 + g * 8]);
    const int nc = wv * 16 + ml;
    v16bf bfrag = cat16(*(const v8bf*)&sB[nc][g * 8],
                        *(const v8bf*)&sB[nc][16 + g * 8]);

    acc0 = wmma_bf16(a0, bfrag, acc0);
    acc1 = wmma_bf16(a1, bfrag, acc1);
    __syncthreads();
  }

  // epilogue: weight, repack in LDS, b128 scatter by slot
  const int colLoc = wv * 16 + ml;
#pragma unroll
  for (int sub = 0; sub < 2; ++sub) {
#pragma unroll
    for (int r = 0; r < 8; ++r) {
      int M = sub * 16 + r + g * 8;
      sOutF[M][colLoc] = (sub ? acc1[r] : acc0[r]) * sW[M];
    }
  }
  __syncthreads();
  for (int idx = tid; idx < 32 * 32; idx += 256) {   // 32 rows x 32 16B-chunks
    int row = idx >> 5, ch = idx & 31;
    int slot = sSlot[row];
    if (slot >= 0)
      *(float4*)(partial + (size_t)slot * DDIM + n0 + ch * 4) =
          *(const float4*)&sOutF[row][ch * 4];
  }
}

__global__ void moe_combine(const float* __restrict__ partial, float* __restrict__ out) {
  size_t i = (size_t)blockIdx.x * blockDim.x + threadIdx.x;   // i = t*D + d
  size_t t = i >> 10;
  size_t d = i & 1023;
  out[i] = partial[(2 * t) * DDIM + d] + partial[(2 * t + 1) * DDIM + d];
}

// ---------------- launch ----------------
extern "C" void kernel_launch(void* const* d_in, const int* in_sizes, int n_in,
                              void* d_out, int out_size, void* d_ws, size_t ws_size,
                              hipStream_t stream) {
  const float* x  = (const float*)d_in[0];   // [B,T,D]
  const float* Wr = (const float*)d_in[1];   // [D,E]
  const float* W1 = (const float*)d_in[2];   // [E,D,F]
  const float* W3 = (const float*)d_in[3];   // [E,D,F]
  const float* W2 = (const float*)d_in[4];   // [E,F,D]
  float* out = (float*)d_out;                // [B,T,D] floats + 1 loss scalar

  char* ws = (char*)d_ws;
  size_t off = 0;
  auto take = [&](size_t bytes) { size_t o = off; off = (off + bytes + 255) & ~(size_t)255; return o; };
  __bf16* xb     = (__bf16*)(ws + take((size_t)NTOK * DDIM * 2));           // 8 MB
  float*  probs  = (float*) (ws + take((size_t)NTOK * NEXP * 4));           // 128 KB
  int*    cnt    = (int*)   (ws + take(NEXP * 4));
  int*    offs   = (int*)   (ws + take((NEXP + 1) * 4));
  int*    tok    = (int*)   (ws + take((size_t)NEXP * NTOK * 4));           // 128 KB
  float*  wl     = (float*) (ws + take((size_t)NEXP * NTOK * 4));           // 128 KB
  __bf16* hbuf   = (__bf16*)(ws + take((size_t)NTOK * TOPK * FDIM * 2));    // 64 MB
  float*  part   = (float*) (ws + take((size_t)NTOK * TOPK * DDIM * 4));    // 32 MB
  (void)ws_size; (void)in_sizes; (void)n_in; (void)out_size;

  moe_init<<<1, 32, 0, stream>>>(cnt);
  moe_cvt_bf16<<<(NTOK * DDIM) / 256, 256, 0, stream>>>(x, xb);
  moe_router<<<NTOK / 8, 256, 0, stream>>>(x, Wr, probs, cnt, tok, wl);
  moe_offsets<<<1, 32, 0, stream>>>(cnt, offs);
  moe_loss<<<1, 256, 0, stream>>>(probs, out + (size_t)NTOK * DDIM);

  dim3 g1(FDIM / 128, NTOK / 32, NEXP);
  moe_gateup<<<g1, 256, 0, stream>>>(xb, W1, W3, cnt, offs, tok, hbuf);

  dim3 g2(DDIM / 128, NTOK / 32, NEXP);
  moe_down<<<g2, 256, 0, stream>>>(hbuf, W2, cnt, offs, tok, wl, part);

  moe_combine<<<(NTOK * DDIM) / 256, 256, 0, stream>>>(part, out);
}
